// Evoformer_62337155334169
// MI455X (gfx1250) — matrix operand order repaired and checked by reading, not compiled, not run on hardware
//
#include <hip/hip_runtime.h>

typedef __attribute__((ext_vector_type(16))) __bf16 v16bf;
typedef __attribute__((ext_vector_type(8)))  float  v8f;

#define WMMA_BF16(a,b,c) __builtin_amdgcn_wmma_f32_16x16x32_bf16(false,(a),false,(b),(short)0,(c),false,false)

static __device__ __forceinline__ float sigm(float x) { return 1.f / (1.f + __expf(-x)); }

// gfx1250 async global->LDS copy (ASYNCcnt), 16B per lane.
static __device__ __forceinline__ void async_b128(unsigned lds_off, const void* gaddr) {
  asm volatile("global_load_async_to_lds_b128 %0, %1, off" :: "v"(lds_off), "v"(gaddr) : "memory");
}
static __device__ __forceinline__ void async_wait() {
  asm volatile("s_wait_asynccnt 0x0" ::: "memory");
}

// ---------------------------------------------------------------------------
// Async-staged GEMM: D = act(A @ B) [or +=].  A,B bf16 row-major, D f32/bf16.
// Block tile 128x64, 8 waves, each wave 2x2 16x16 tiles -> 4 WMMA per k-step.
// Requires M%128==0, N%64==0, K%32==0.  LDS staged via async b128 copies.
// ---------------------------------------------------------------------------
template<int ACT, bool ACC, typename OutT>
__global__ __launch_bounds__(256) void gemm_async(
    const __bf16* __restrict__ A, const __bf16* __restrict__ B, OutT* __restrict__ D,
    int M, int N, int K, int lda, int ldb, int ldc)
{
  extern __shared__ char smem[];
  const unsigned BBASE = 128 * 80;               // As: 128 rows * 80B (40 bf16)
  __bf16* As = (__bf16*)smem;                    // [128][40] bf16
  __bf16* Bs = (__bf16*)(smem + BBASE);          // [32][72] bf16 (144B rows)

  const int tid = threadIdx.x, wave = tid >> 5, lane = tid & 31;
  const int half = lane >> 4, l15 = lane & 15;
  const int m0 = blockIdx.y * 128, n0 = blockIdx.x * 64;
  const int mw = wave & 3, nw = wave >> 2;       // 4 m-groups x 2 n-groups

  v8f acc[2][2] = {{{}, {}}, {{}, {}}};

  for (int kb = 0; kb < K; kb += 32) {
    { // A tile: 128 rows x 32 bf16 = 512 b128 chunks (2/thread)
      int i0 = tid, i1 = tid + 256;
      int r0 = i0 >> 2, c0 = i0 & 3, r1 = i1 >> 2, c1 = i1 & 3;
      async_b128((unsigned)r0 * 80u + c0 * 16u, A + (long)(m0 + r0) * lda + kb + c0 * 8);
      async_b128((unsigned)r1 * 80u + c1 * 16u, A + (long)(m0 + r1) * lda + kb + c1 * 8);
    }
    { // B tile: 32 rows x 64 bf16 = 256 chunks (1/thread)
      int r = tid >> 3, c = tid & 7;
      int gn = n0 + c * 8;
      const __bf16* src = (gn + 8 <= N) ? (B + (long)(kb + r) * ldb + gn) : B;
      async_b128(BBASE + (unsigned)r * 144u + c * 16u, src);
    }
    async_wait();
    __syncthreads();

    v16bf af[2], bf[2];
#pragma unroll
    for (int a = 0; a < 2; ++a) {
      int arow = (mw * 2 + a) * 16 + l15;
#pragma unroll
      for (int v = 0; v < 8; ++v) {
        int p = (v >> 2) * 16 + half * 8 + (v & 3) * 2;
        af[a][2 * v]     = As[arow * 40 + p];
        af[a][2 * v + 1] = As[arow * 40 + p + 1];
      }
    }
#pragma unroll
    for (int b = 0; b < 2; ++b) {
      int col = (nw * 2 + b) * 16 + l15;
#pragma unroll
      for (int v = 0; v < 8; ++v) {
        int kr = (v >> 2) * 16 + half * 8 + (v & 3) * 2;
        bf[b][2 * v]     = Bs[kr * 72 + col];
        bf[b][2 * v + 1] = Bs[(kr + 1) * 72 + col];
      }
    }
#pragma unroll
    for (int a = 0; a < 2; ++a)
#pragma unroll
      for (int b = 0; b < 2; ++b)
        acc[a][b] = WMMA_BF16(af[a], bf[b], acc[a][b]);
    __syncthreads();
  }

#pragma unroll
  for (int a = 0; a < 2; ++a) {
    int rbase = m0 + (mw * 2 + a) * 16 + half * 8;
#pragma unroll
    for (int b = 0; b < 2; ++b) {
      int col = n0 + (nw * 2 + b) * 16 + l15;
      if (col >= N) continue;
#pragma unroll
      for (int i = 0; i < 8; ++i) {
        float r = acc[a][b][i];
        if (ACT == 1) r = sigm(r); else if (ACT == 2) r = fmaxf(r, 0.f);
        long o = (long)(rbase + i) * ldc + col;
        if (ACC) D[o] += r; else D[o] = (OutT)r;
      }
    }
  }
}

// ---------------------------------------------------------------------------
// Guarded sync-load GEMM (bf16 in): handles transA/transB (triangle-mult
// batched einsums) and small N (bias projections).  64x64 tile, 8 waves.
// ---------------------------------------------------------------------------
template<int ACT, bool ACC, bool TA, bool TB, typename OutT>
__global__ __launch_bounds__(256) void gemm_sync(
    const __bf16* __restrict__ A, const __bf16* __restrict__ B, OutT* __restrict__ D,
    int M, int N, int K, int lda, int ldb, int ldc, long sA, long sB, long sD)
{
  const int bz = blockIdx.z;
  A += (long)bz * sA; B += (long)bz * sB; D += (long)bz * sD;
  __shared__ __bf16 As[64][36];
  __shared__ __bf16 Bs[32][68];
  const int tid = threadIdx.x, wave = tid >> 5, lane = tid & 31;
  const int half = lane >> 4, l15 = lane & 15;
  const int m0 = blockIdx.y * 64, n0 = blockIdx.x * 64;
  const int mi = wave & 3, nb = (wave >> 2) * 2;
  v8f acc0 = {}, acc1 = {};

  for (int kb = 0; kb < K; kb += 32) {
    for (int i = tid; i < 64 * 32; i += 256) {
      int r = i >> 5, c = i & 31;
      int gm = m0 + r, gk = kb + c;
      __bf16 v = (__bf16)0.f;
      if (gm < M && gk < K) v = TA ? A[(long)gk * lda + gm] : A[(long)gm * lda + gk];
      As[r][c] = v;
    }
    for (int i = tid; i < 32 * 64; i += 256) {
      int r = i >> 6, c = i & 63;
      int gk = kb + r, gn = n0 + c;
      __bf16 v = (__bf16)0.f;
      if (gk < K && gn < N) v = TB ? B[(long)gn * ldb + gk] : B[(long)gk * ldb + gn];
      Bs[r][c] = v;
    }
    __syncthreads();
    v16bf af;
    const int arow = mi * 16 + l15;
#pragma unroll
    for (int v = 0; v < 8; ++v) {
      int p = (v >> 2) * 16 + half * 8 + (v & 3) * 2;
      af[2 * v] = As[arow][p]; af[2 * v + 1] = As[arow][p + 1];
    }
    v16bf bf0, bf1;
    const int bc0 = nb * 16 + l15, bc1 = bc0 + 16;
#pragma unroll
    for (int v = 0; v < 8; ++v) {
      int kr = (v >> 2) * 16 + half * 8 + (v & 3) * 2;
      bf0[2 * v] = Bs[kr][bc0]; bf0[2 * v + 1] = Bs[kr + 1][bc0];
      bf1[2 * v] = Bs[kr][bc1]; bf1[2 * v + 1] = Bs[kr + 1][bc1];
    }
    acc0 = WMMA_BF16(af, bf0, acc0);
    acc1 = WMMA_BF16(af, bf1, acc1);
    __syncthreads();
  }
#pragma unroll
  for (int i = 0; i < 8; ++i) {
    int row = m0 + mi * 16 + half * 8 + i;
    if (row >= M) continue;
    float r0 = acc0[i], r1 = acc1[i];
    if (ACT == 1) { r0 = sigm(r0); r1 = sigm(r1); }
    else if (ACT == 2) { r0 = fmaxf(r0, 0.f); r1 = fmaxf(r1, 0.f); }
    long base = (long)row * ldc;
    int c0 = n0 + nb * 16 + l15, c1 = c0 + 16;
    if (c0 < N) { if (ACC) D[base + c0] += r0; else D[base + c0] = (OutT)r0; }
    if (c1 < N) { if (ACC) D[base + c1] += r1; else D[base + c1] = (OutT)r1; }
  }
}

// ---------------------------------------------------------------------------
// Fused gated attention, bf16 operands, async K/V/Q staging, f32 softmax.
// sim = softmax(qk*scale); sim += bias (post-softmax per reference);
// O = sim@V * gate.  biasMode: 0 none, 1 [(q*L+k)*H+h], 2 [(s*L+k)*H+h].
// Dynamic LDS layout (bytes): K 0..18431, V ..36863, Q ..38015(=36864+1152),
// S f32 ..54911, P bf16 ..63359.
// ---------------------------------------------------------------------------
__global__ __launch_bounds__(256) void attn_wmma(
    const __bf16* __restrict__ qkv, const __bf16* __restrict__ gate,
    const float* __restrict__ bias, __bf16* __restrict__ outp,
    int L, int H, int biasMode)
{
  const float SCALE = 0.17677669529663687f; // 1/sqrt(32)
  extern __shared__ char smem[];
  const unsigned KB = 0, VB = 18432, QB = 36864, SB = 38016, PB = 54912;
  __bf16* Ks = (__bf16*)(smem + KB);   // [256][36] bf16 (72B rows)
  __bf16* Vs = (__bf16*)(smem + VB);
  __bf16* Qs = (__bf16*)(smem + QB);   // [16][36]
  float*  Ss = (float*)(smem + SB);    // [16][264]
  __bf16* Ps = (__bf16*)(smem + PB);   // [16][264]

  const int s = blockIdx.x, h = blockIdx.y;
  const int HC = H * 32, QW = 3 * HC;
  const int tid = threadIdx.x, wave = tid >> 5, lane = tid & 31;
  const int half = lane >> 4, l15 = lane & 15;

  for (int i = tid; i < L * 8; i += 256) {   // K & V: L rows x 4 chunks each
    int r = i >> 3, rem = i & 7, which = rem >> 2, ch = rem & 3;
    const __bf16* src = qkv + (long)(s * L + r) * QW + (which ? 2 * HC : HC) + h * 32 + ch * 8;
    async_b128((which ? VB : KB) + (unsigned)r * 72u + ch * 16u, src);
  }

  const int nkt = L >> 4, nkc = L >> 5;
  for (int q0 = 0; q0 < L; q0 += 16) {
    if (tid < 64) {
      int r = tid >> 2, ch = tid & 3;
      async_b128(QB + (unsigned)r * 72u + ch * 16u,
                 qkv + (long)(s * L + q0 + r) * QW + h * 32 + ch * 8);
    }
    async_wait();
    __syncthreads();

    v16bf af;
#pragma unroll
    for (int v = 0; v < 8; ++v) {
      int p = (v >> 2) * 16 + half * 8 + (v & 3) * 2;
      af[2 * v] = Qs[l15 * 36 + p]; af[2 * v + 1] = Qs[l15 * 36 + p + 1];
    }
    for (int t = wave; t < nkt; t += 8) {
      v16bf bf;
      int col = t * 16 + l15;
#pragma unroll
      for (int v = 0; v < 8; ++v) {
        int kr = (v >> 2) * 16 + half * 8 + (v & 3) * 2;
        bf[2 * v] = Ks[col * 36 + kr]; bf[2 * v + 1] = Ks[col * 36 + kr + 1];
      }
      v8f c = {};
      c = WMMA_BF16(af, bf, c);
#pragma unroll
      for (int i = 0; i < 8; ++i)
        Ss[(half * 8 + i) * 264 + t * 16 + l15] = c[i] * SCALE;
    }
    __syncthreads();

    if (tid < 16) {
      int row = tid, qg = q0 + row;
      float m = -1e30f;
      for (int j = 0; j < L; ++j) m = fmaxf(m, Ss[row * 264 + j]);
      float sum = 0.f;
      for (int j = 0; j < L; ++j) { float e = __expf(Ss[row * 264 + j] - m); Ss[row * 264 + j] = e; sum += e; }
      float inv = 1.f / sum;
      for (int j = 0; j < L; ++j) {
        float v = Ss[row * 264 + j] * inv;
        if (biasMode == 1)      v += bias[((long)qg * L + j) * H + h];
        else if (biasMode == 2) v += bias[((long)s  * L + j) * H + h];
        Ps[row * 264 + j] = (__bf16)v;
      }
    }
    __syncthreads();

    if (wave < 2) {
      int ni = wave;
      v8f acc = {};
      for (int kc = 0; kc < nkc; ++kc) {
        v16bf pf, vf;
#pragma unroll
        for (int v = 0; v < 8; ++v) {
          int p = kc * 32 + (v >> 2) * 16 + half * 8 + (v & 3) * 2;
          pf[2 * v] = Ps[l15 * 264 + p]; pf[2 * v + 1] = Ps[l15 * 264 + p + 1];
          int kr = kc * 32 + (v >> 2) * 16 + half * 8 + (v & 3) * 2;
          vf[2 * v] = Vs[kr * 36 + ni * 16 + l15]; vf[2 * v + 1] = Vs[(kr + 1) * 36 + ni * 16 + l15];
        }
        acc = WMMA_BF16(pf, vf, acc);
      }
#pragma unroll
      for (int i = 0; i < 8; ++i) {
        int q = q0 + half * 8 + i;
        long o = (long)(s * L + q) * HC + h * 32 + ni * 16 + l15;
        outp[o] = (__bf16)(acc[i] * (float)gate[o]);
      }
    }
    __syncthreads();
  }
}

// ---------------------------------------------------------------------------
// Fused outer-product-mean: per (r,t), M32 = a[:,r,:]^T @ b[:,t,:] via WMMA
// over S=128, then pair[r,t,:] += M32.flat @ W.  Async staging.
// ---------------------------------------------------------------------------
__global__ __launch_bounds__(128) void outer_wmma(
    const __bf16* __restrict__ a, const __bf16* __restrict__ b,
    const float* __restrict__ W, float* __restrict__ pair,
    int S_, int R_, int CZ_)
{
  extern __shared__ char smem[];
  __bf16* As = (__bf16*)smem;             // [128][36] bf16
  __bf16* Bs = (__bf16*)(smem + 9216);
  float*  Ms = (float*)(smem + 18432);    // [32][33]
  const int r = blockIdx.x, t = blockIdx.y, tid = threadIdx.x;

  for (int i = tid; i < 1024; i += 128) {
    int which = i >> 9, idx = i & 511, si = idx >> 2, ch = idx & 3;
    const __bf16* src = (which ? b + ((long)si * R_ + t) * 32 : a + ((long)si * R_ + r) * 32) + ch * 8;
    async_b128((which ? 9216u : 0u) + (unsigned)si * 72u + ch * 16u, src);
  }
  async_wait();
  __syncthreads();

  const int wave = tid >> 5, lane = tid & 31, half = lane >> 4, l15 = lane & 15;
  const int mi = wave >> 1, ni = wave & 1;
  v8f acc = {};
  for (int kc = 0; kc < S_; kc += 32) {
    v16bf af, bf;
#pragma unroll
    for (int v = 0; v < 8; ++v) {
      int p = kc + (v >> 2) * 16 + half * 8 + (v & 3) * 2;
      af[2 * v] = As[p * 36 + mi * 16 + l15]; af[2 * v + 1] = As[(p + 1) * 36 + mi * 16 + l15];
      int kr = kc + (v >> 2) * 16 + half * 8 + (v & 3) * 2;
      bf[2 * v] = Bs[kr * 36 + ni * 16 + l15]; bf[2 * v + 1] = Bs[(kr + 1) * 36 + ni * 16 + l15];
    }
    acc = WMMA_BF16(af, bf, acc);
  }
#pragma unroll
  for (int i = 0; i < 8; ++i)
    Ms[(mi * 16 + half * 8 + i) * 33 + ni * 16 + l15] = acc[i];
  __syncthreads();
  if (tid < CZ_) {
    float av = 0.f;
    for (int ce = 0; ce < 1024; ++ce)
      av += Ms[(ce >> 5) * 33 + (ce & 31)] * W[(long)ce * CZ_ + tid];
    pair[((long)r * R_ + t) * CZ_ + tid] += av;
  }
}

// ---------------------------------------------------------------------------
// LayerNorm over last dim D (f32 math), bf16 output for WMMA consumers.
// ---------------------------------------------------------------------------
template<typename InT>
__global__ __launch_bounds__(128) void ln_kernel(
    const InT* __restrict__ x, const float* __restrict__ w,
    const float* __restrict__ b, __bf16* __restrict__ y, int D)
{
  long row = blockIdx.x;
  const InT* xr = x + row * D;
  __bf16* yr = y + row * D;
  __shared__ float red[128];
  float s = 0.f;
  for (int i = threadIdx.x; i < D; i += 128) s += (float)xr[i];
  red[threadIdx.x] = s; __syncthreads();
  for (int o = 64; o > 0; o >>= 1) { if (threadIdx.x < o) red[threadIdx.x] += red[threadIdx.x + o]; __syncthreads(); }
  float mean = red[0] / D; __syncthreads();
  float s2 = 0.f;
  for (int i = threadIdx.x; i < D; i += 128) { float d = (float)xr[i] - mean; s2 += d * d; }
  red[threadIdx.x] = s2; __syncthreads();
  for (int o = 64; o > 0; o >>= 1) { if (threadIdx.x < o) red[threadIdx.x] += red[threadIdx.x + o]; __syncthreads(); }
  float inv = rsqrtf(red[0] / D + 1e-5f);
  for (int i = threadIdx.x; i < D; i += 128)
    yr[i] = (__bf16)(((float)xr[i] - mean) * inv * w[i] + b[i]);
}

// out[b][a][c] (+)= in[a][b][c]
template<typename TI, typename TO, bool ADD>
__global__ void transpose_sr(const TI* __restrict__ in, TO* __restrict__ out,
                             int A, int B, int Cc)
{
  long idx = (long)blockIdx.x * 256 + threadIdx.x;
  long total = (long)A * B * Cc;
  if (idx >= total) return;
  int c = (int)(idx % Cc); long t = idx / Cc;
  int b = (int)(t % B); int a = (int)(t / B);
  float v = (float)in[((long)a * B + b) * Cc + c];
  long o = ((long)b * A + a) * Cc + c;
  if (ADD) out[o] += v; else out[o] = (TO)v;
}

// forward: out[(k*D0+i)*D1+j] = in[(i*D1+j)*Kc+k]; INV: reverse
template<bool INV>
__global__ void permute_kf(const __bf16* __restrict__ in, __bf16* __restrict__ out,
                           int D0, int D1, int Kc)
{
  long idx = (long)blockIdx.x * 256 + threadIdx.x;
  long total = (long)D0 * D1 * Kc;
  if (idx >= total) return;
  int k = (int)(idx % Kc); long t = idx / Kc;
  int j = (int)(t % D1); int i = (int)(t / D1);
  long a = ((long)i * D1 + j) * Kc + k;
  long b = ((long)k * D0 + i) * D1 + j;
  if (INV) out[a] = in[b]; else out[b] = in[a];
}

__global__ void ew_mul(__bf16* a, const __bf16* __restrict__ b, long n)
{ long i = (long)blockIdx.x * 256 + threadIdx.x; if (i < n) a[i] = (__bf16)((float)a[i] * (float)b[i]); }

__global__ void ew_fma(float* d, const __bf16* __restrict__ a, const __bf16* __restrict__ b, long n)
{ long i = (long)blockIdx.x * 256 + threadIdx.x; if (i < n) d[i] += (float)a[i] * (float)b[i]; }

__global__ void cvtk(const float* __restrict__ s, __bf16* __restrict__ d, long n)
{ long i = (long)blockIdx.x * 256 + threadIdx.x; if (i < n) d[i] = (__bf16)s[i]; }

// ---------------------------------------------------------------------------
// Host-side dispatch
// ---------------------------------------------------------------------------
static const size_t GEMM_SH = 128 * 80 + 32 * 144;

static inline void Gb(hipStream_t st, const __bf16* A, const __bf16* B, __bf16* D,
                      int M, int N, int K, int act)
{
  if (N % 64 == 0 && M % 128 == 0) {
    dim3 g(N / 64, M / 128);
    if (act == 1)      gemm_async<1, false, __bf16><<<g, 256, GEMM_SH, st>>>(A, B, D, M, N, K, K, N, N);
    else if (act == 2) gemm_async<2, false, __bf16><<<g, 256, GEMM_SH, st>>>(A, B, D, M, N, K, K, N, N);
    else               gemm_async<0, false, __bf16><<<g, 256, GEMM_SH, st>>>(A, B, D, M, N, K, K, N, N);
  } else {
    dim3 g((N + 63) / 64, (M + 63) / 64);
    gemm_sync<0, false, false, false, __bf16><<<g, 256, 0, st>>>(A, B, D, M, N, K, K, N, N, 0, 0, 0);
  }
}
static inline void Gacc(hipStream_t st, const __bf16* A, const __bf16* B, float* D,
                        int M, int N, int K)
{
  dim3 g(N / 64, M / 128);
  gemm_async<0, true, float><<<g, 256, GEMM_SH, st>>>(A, B, D, M, N, K, K, N, N);
}
static inline void Gbias(hipStream_t st, const __bf16* A, const __bf16* B, float* D,
                         int M, int N, int K)
{
  dim3 g((N + 63) / 64, (M + 63) / 64);
  gemm_sync<0, false, false, false, float><<<g, 256, 0, st>>>(A, B, D, M, N, K, K, N, N, 0, 0, 0);
}

extern "C" void kernel_launch(void* const* d_in, const int* in_sizes, int n_in,
                              void* d_out, int out_size, void* d_ws, size_t ws_size,
                              hipStream_t stream)
{
  (void)in_sizes; (void)n_in; (void)out_size; (void)ws_size;
  const int S = 128, R = 256, CM = 256, CZ = 128;
  const int Mm = S * R;        // 32768
  const int Mz = R * R;        // 65536
  const size_t M8 = 8388608;   // elements per residual tensor
  const size_t B16 = 16777216; // bytes of one bf16 M8 buffer

  char* wsb = (char*)d_ws;
  __bf16* xbuf  = (__bf16*)(wsb + 0 * B16);
  __bf16* xbuf2 = (__bf16*)(wsb + 1 * B16);
  __bf16* qkvb  = (__bf16*)(wsb + 2 * B16);   // 3*B16
  __bf16* gateb = (__bf16*)(wsb + 5 * B16);
  __bf16* attb  = (__bf16*)(wsb + 6 * B16);
  __bf16* t1b   = (__bf16*)(wsb + 7 * B16);   // 4*B16
  __bf16* abuf  = (__bf16*)(wsb + 11 * B16);
  __bf16* bbuf  = (__bf16*)(wsb + 12 * B16);
  __bf16* akb   = (__bf16*)(wsb + 13 * B16);
  __bf16* bkb   = (__bf16*)(wsb + 14 * B16);
  __bf16* pkb   = (__bf16*)(wsb + 15 * B16);
  __bf16* pb    = (__bf16*)(wsb + 16 * B16);
  __bf16* pnb   = (__bf16*)(wsb + 17 * B16);
  __bf16* gb    = (__bf16*)(wsb + 18 * B16);
  __bf16* gb2   = (__bf16*)(wsb + 19 * B16);
  __bf16* tb    = (__bf16*)(wsb + 20 * B16);
  float*  biasb = (float*)(wsb + 21 * B16);   // 2MB
  __bf16* wbuf  = (__bf16*)(wsb + 21 * B16 + 2097152);

  float* msa  = (float*)d_out;
  float* pair = msa + M8;
  const float* const* in = (const float* const*)d_in;

  hipMemcpyAsync(msa,  in[0], M8 * sizeof(float), hipMemcpyDeviceToDevice, stream);
  hipMemcpyAsync(pair, in[1], M8 * sizeof(float), hipMemcpyDeviceToDevice, stream);

  // --- convert all GEMM weight operands to bf16 (deterministic every call) ---
  __bf16* wp = wbuf;
  auto CW = [&](int idx, long n) {
    __bf16* p = wp; wp += n;
    cvtk<<<dim3((unsigned)((n + 255) / 256)), 256, 0, stream>>>(in[idx], p, n);
    return p;
  };
  __bf16 *w6 = CW(6, 256 * 256),  *w7 = CW(7, 256 * 768),  *w8 = CW(8, 256 * 256),  *w9 = CW(9, 128 * 8);
  __bf16 *w12 = CW(12, 256 * 256), *w13 = CW(13, 256 * 768), *w14 = CW(14, 256 * 256);
  __bf16 *w17 = CW(17, 256 * 1024), *w18 = CW(18, 1024 * 256);
  __bf16 *w21 = CW(21, 256 * 32),  *w22 = CW(22, 256 * 32);
  __bf16 *wtm[2][6];
  for (int d = 0; d < 2; ++d)
    for (int k = 0; k < 6; ++k) wtm[d][k] = CW(28 + d * 10 + k, 128 * 128);
  __bf16 *w46 = CW(46, 128 * 128), *w47 = CW(47, 128 * 384), *w48 = CW(48, 128 * 128), *w49 = CW(49, 128 * 4);
  __bf16 *w52 = CW(52, 128 * 128), *w53 = CW(53, 128 * 384), *w54 = CW(54, 128 * 128), *w55 = CW(55, 128 * 4);
  __bf16 *w58 = CW(58, 128 * 512), *w59 = CW(59, 512 * 128);

  const size_t ATT_SH = 63360, OUT_SH = 22656;
  long nE = (long)Mz * 128;
  dim3 eb((unsigned)((nE + 255) / 256));
  dim3 em((unsigned)(((long)Mm * 256 + 255) / 256));

  // ---- Row attention with pair bias ----
  ln_kernel<float><<<Mm, 128, 0, stream>>>(msa,  in[2], in[3], xbuf,  CM);
  ln_kernel<float><<<Mz, 128, 0, stream>>>(pair, in[4], in[5], xbuf2, CZ);
  Gbias(stream, xbuf2, w9, biasb, Mz, 8, CZ);
  Gb(stream, xbuf, w6, gateb, Mm, 256, 256, 1);
  Gb(stream, xbuf, w7, qkvb,  Mm, 768, 256, 0);
  attn_wmma<<<dim3(S, 8), 256, ATT_SH, stream>>>(qkvb, gateb, biasb, attb, 256, 8, 1);
  Gacc(stream, attb, w8, msa, Mm, 256, 256);

  // ---- Column attention (transposed MSA) ----
  ln_kernel<float><<<Mm, 128, 0, stream>>>(msa, in[10], in[11], xbuf, CM);
  transpose_sr<__bf16, __bf16, false><<<em, 256, 0, stream>>>(xbuf, xbuf2, S, R, CM);
  Gb(stream, xbuf2, w12, gateb, Mm, 256, 256, 1);
  Gb(stream, xbuf2, w13, qkvb,  Mm, 768, 256, 0);
  attn_wmma<<<dim3(R, 8), 256, ATT_SH, stream>>>(qkvb, gateb, nullptr, attb, 128, 8, 0);
  Gb(stream, attb, w14, xbuf, Mm, 256, 256, 0);
  transpose_sr<__bf16, float, true><<<em, 256, 0, stream>>>(xbuf, msa, R, S, CM);

  // ---- MSA transition ----
  ln_kernel<float><<<Mm, 128, 0, stream>>>(msa, in[15], in[16], xbuf, CM);
  Gb(stream, xbuf, w17, t1b, Mm, 1024, 256, 2);
  Gacc(stream, t1b, w18, msa, Mm, 256, 1024);

  // ---- Outer product mean ----
  ln_kernel<float><<<Mm, 128, 0, stream>>>(msa, in[19], in[20], xbuf, CM);
  Gb(stream, xbuf, w21, abuf, Mm, 32, 256, 0);
  Gb(stream, xbuf, w22, bbuf, Mm, 32, 256, 0);
  outer_wmma<<<dim3(R, R), 128, OUT_SH, stream>>>(abuf, bbuf, in[23], pair, S, R, CZ);

  // ---- Triangle multiplications (outgoing, incoming) ----
  for (int dir = 0; dir < 2; ++dir) {
    const float *n1w = in[24 + dir * 10], *n1b = in[25 + dir * 10];
    const float *n2w = in[26 + dir * 10], *n2b = in[27 + dir * 10];
    ln_kernel<float><<<Mz, 128, 0, stream>>>(pair, n1w, n1b, xbuf, CZ);
    Gb(stream, xbuf, wtm[dir][0], abuf, Mz, 128, 128, 1);
    Gb(stream, xbuf, wtm[dir][1], tb,   Mz, 128, 128, 0);
    ew_mul<<<eb, 256, 0, stream>>>(abuf, tb, nE);
    Gb(stream, xbuf, wtm[dir][2], bbuf, Mz, 128, 128, 1);
    Gb(stream, xbuf, wtm[dir][3], tb,   Mz, 128, 128, 0);
    ew_mul<<<eb, 256, 0, stream>>>(bbuf, tb, nE);
    permute_kf<false><<<eb, 256, 0, stream>>>(abuf, akb, 256, 256, 128);
    permute_kf<false><<<eb, 256, 0, stream>>>(bbuf, bkb, 256, 256, 128);
    dim3 bg(4, 4, 128);
    if (dir == 0)  // P[s,t] = sum_r a[s,r] b[t,r]  (transB)
      gemm_sync<0, false, false, true, __bf16><<<bg, 256, 0, stream>>>(
          akb, bkb, pkb, 256, 256, 256, 256, 256, 256, 65536, 65536, 65536);
    else           // P[s,t] = sum_r a[r,s] b[r,t]  (transA)
      gemm_sync<0, false, true, false, __bf16><<<bg, 256, 0, stream>>>(
          akb, bkb, pkb, 256, 256, 256, 256, 256, 256, 65536, 65536, 65536);
    permute_kf<true><<<eb, 256, 0, stream>>>(pkb, pb, 256, 256, 128);
    ln_kernel<__bf16><<<Mz, 128, 0, stream>>>(pb, n2w, n2b, pnb, 128);
    Gb(stream, xbuf, wtm[dir][5], gb, Mz, 128, 128, 1);
    __bf16* guse = gb;
    if (dir == 1) {
      transpose_sr<__bf16, __bf16, false><<<eb, 256, 0, stream>>>(gb, gb2, 256, 256, 128);
      guse = gb2;
    }
    Gb(stream, pnb, wtm[dir][4], tb, Mz, 128, 128, 0);
    ew_fma<<<eb, 256, 0, stream>>>(pair, tb, guse, nE);
  }

  // ---- Triangle attention (starting node) ----
  ln_kernel<float><<<Mz, 128, 0, stream>>>(pair, in[44], in[45], xbuf, CZ);
  Gbias(stream, xbuf, w49, biasb, Mz, 4, 128);
  Gb(stream, xbuf, w46, gateb, Mz, 128, 128, 1);
  Gb(stream, xbuf, w47, qkvb,  Mz, 384, 128, 0);
  attn_wmma<<<dim3(R, 4), 256, ATT_SH, stream>>>(qkvb, gateb, biasb, attb, 256, 4, 2);
  Gacc(stream, attb, w48, pair, Mz, 128, 128);

  // ---- Triangle attention (ending node, transposed) ----
  ln_kernel<float><<<Mz, 128, 0, stream>>>(pair, in[50], in[51], xbuf, CZ);
  transpose_sr<__bf16, __bf16, false><<<eb, 256, 0, stream>>>(xbuf, xbuf2, 256, 256, 128);
  Gbias(stream, xbuf2, w55, biasb, Mz, 4, 128);
  Gb(stream, xbuf2, w52, gateb, Mz, 128, 128, 1);
  Gb(stream, xbuf2, w53, qkvb,  Mz, 384, 128, 0);
  attn_wmma<<<dim3(R, 4), 256, ATT_SH, stream>>>(qkvb, gateb, biasb, attb, 256, 4, 2);
  Gb(stream, attb, w54, xbuf, Mz, 128, 128, 0);
  transpose_sr<__bf16, float, true><<<eb, 256, 0, stream>>>(xbuf, pair, 256, 256, 128);

  // ---- Pair transition ----
  ln_kernel<float><<<Mz, 128, 0, stream>>>(pair, in[56], in[57], xbuf, CZ);
  Gb(stream, xbuf, w58, t1b, Mz, 512, 128, 2);
  Gacc(stream, t1b, w59, pair, Mz, 128, 512);
}
